// BRM_59674275611311
// MI455X (gfx1250) — matrix operand order repaired and checked by reading, not compiled, not run on hardware
//
#include <hip/hip_runtime.h>
#include <hip/hip_bf16.h>
#include <math.h>

#define NGRAPH 128
#define EPS_BN 1e-5f
#define EPS_LN 1e-5f

typedef __bf16 bf16;
typedef bf16 v16bf __attribute__((ext_vector_type(16)));
typedef float v8f  __attribute__((ext_vector_type(8)));

__device__ __forceinline__ float reluf(float x) { return fmaxf(x, 0.f); }

__device__ __forceinline__ void atomicMaxF(float* a, float v) {
    // IEEE ordering trick: non-negative -> int max, negative -> uint min
    if (v >= 0.f) atomicMax((int*)a, __float_as_int(v));
    else          atomicMin((unsigned int*)a, __float_as_uint(v));
}

// ---------------------------------------------------------------------------
// WMMA GEMM: Y[M,64] = X[M,K] @ W[K,64] + bias, optional per-row LayerNorm.
// bf16 operands, f32 accumulate via V_WMMA_F32_16X16X32_BF16.
// Block = 256 threads = 8 waves; each wave computes a 16x64 tile (128 rows/blk).
// Weights staged transposed+zero-padded in LDS so all fragment loads are
// branch-free, vectorized (global_load_b128 / merged ds_load_b128).
// ---------------------------------------------------------------------------
template<int KDIM, bool DO_LN>
__global__ __launch_bounds__(256) void k_gemm(
    const float* __restrict__ X, int M,
    const float* __restrict__ W,      // [KDIM,64] row-major
    const float* __restrict__ bias,   // [64]
    const float* __restrict__ lng,    // LN gamma [64] (or null)
    const float* __restrict__ lnb,    // LN beta  [64]
    float* __restrict__ Y)
{
    constexpr int KSTEPS = (KDIM + 31) / 32;
    constexpr int KPAD   = KSTEPS * 32;          // zero-padded K extent
    __shared__ bf16  Wt[64][KPAD];     // transposed weights: Wt[n][k], pad = 0
    __shared__ float tile[8][16][68];  // per-wave staging, padded rows

    const int tid = threadIdx.x;
    for (int idx = tid; idx < 64 * KPAD; idx += 256) {
        const int n = idx / KPAD;
        const int k = idx & (KPAD - 1);
        Wt[n][k] = (k < KDIM) ? (bf16)W[k * 64 + n] : (bf16)0.f;
    }
    __syncthreads();

    const int wave = tid >> 5;
    const int lane = tid & 31;
    const int hsel = lane >> 4;   // 0: lanes 0-15, 1: lanes 16-31
    const int lm   = lane & 15;
    const int row0 = blockIdx.x * 128 + wave * 16;

    v8f acc[4];
    #pragma unroll
    for (int t = 0; t < 4; ++t)
        #pragma unroll
        for (int r = 0; r < 8; ++r) acc[t][r] = 0.f;

    int arow = row0 + lm; if (arow > M - 1) arow = M - 1;   // clamp tail loads
    const float* xr = X + (size_t)arow * KDIM;

    #pragma unroll
    for (int s = 0; s < KSTEPS; ++s) {
        const int k0 = 32 * s;
        // A fragment (16x32 bf16): lane(m=lm): elem e<8 -> k=k0+8*hsel+e,
        //                                     e>=8 -> k=k0+16+8*hsel+(e-8)
        // For KDIM==16 the upper half (e>=8) is entirely k>=16: compile-time pad.
        const float* ap = xr + k0 + 8 * hsel;
        float4 a0 = *reinterpret_cast<const float4*>(ap);
        float4 a1 = *reinterpret_cast<const float4*>(ap + 4);
        float4 a2, a3;
        if (KDIM % 32 == 0) {
            a2 = *reinterpret_cast<const float4*>(ap + 16);
            a3 = *reinterpret_cast<const float4*>(ap + 20);
        } else {
            a2 = make_float4(0.f, 0.f, 0.f, 0.f);
            a3 = a2;
        }
        v16bf a;
        a[0]  = (bf16)a0.x; a[1]  = (bf16)a0.y; a[2]  = (bf16)a0.z; a[3]  = (bf16)a0.w;
        a[4]  = (bf16)a1.x; a[5]  = (bf16)a1.y; a[6]  = (bf16)a1.z; a[7]  = (bf16)a1.w;
        a[8]  = (bf16)a2.x; a[9]  = (bf16)a2.y; a[10] = (bf16)a2.z; a[11] = (bf16)a2.w;
        a[12] = (bf16)a3.x; a[13] = (bf16)a3.y; a[14] = (bf16)a3.z; a[15] = (bf16)a3.w;

        #pragma unroll
        for (int t = 0; t < 4; ++t) {
            // B fragment (32x16 bf16): lane: n = 16t+lm, e<8 -> k = k0 + 8*hsel + e, e>=8 -> k = k0 + 16 + 8*hsel + (e-8).
            // Unconditional: pad region of Wt is zero-filled.
            const int n  = 16 * t + lm;
            const int kb = k0 + 8 * hsel;
            const unsigned int* wrow =
                reinterpret_cast<const unsigned int*>(&Wt[n][0]) + (kb >> 1);
            union { unsigned int u[8]; v16bf v; } bu;
            #pragma unroll
            for (int j = 0; j < 8; ++j) bu.u[j] = wrow[j < 4 ? j : j + 4];
            acc[t] = __builtin_amdgcn_wmma_f32_16x16x32_bf16(
                false, a, false, bu.v, (short)0, acc[t], false, false);
        }
    }

    // Stage C (VGPR r: rows r / r+8, cols = lanes) to LDS with bias added.
    #pragma unroll
    for (int t = 0; t < 4; ++t) {
        const int n = 16 * t + lm;
        const float bb = bias[n];
        #pragma unroll
        for (int r = 0; r < 8; ++r)
            tile[wave][r + 8 * hsel][n] = acc[t][r] + bb;
    }
    __syncthreads();

    // Each lane handles half a row (32 cols), read once as float4s.
    const int orow  = lane & 15;
    const int ocol0 = 32 * hsel;
    float4 vv[8];
    #pragma unroll
    for (int j4 = 0; j4 < 8; ++j4)
        vv[j4] = *reinterpret_cast<const float4*>(&tile[wave][orow][ocol0 + 4 * j4]);

    float mu = 0.f, rstd = 1.f;
    if (DO_LN) {
        float s = 0.f, q = 0.f;
        #pragma unroll
        for (int j4 = 0; j4 < 8; ++j4) {
            s += vv[j4].x + vv[j4].y + vv[j4].z + vv[j4].w;
            q += vv[j4].x * vv[j4].x + vv[j4].y * vv[j4].y
               + vv[j4].z * vv[j4].z + vv[j4].w * vv[j4].w;
        }
        s += __shfl_xor(s, 16, 32);
        q += __shfl_xor(q, 16, 32);
        mu = s * (1.f / 64.f);
        float var = q * (1.f / 64.f) - mu * mu;
        rstd = rsqrtf(var + EPS_LN);
    }
    const int grow = row0 + orow;
    if (grow < M) {
        float* yr = Y + (size_t)grow * 64;
        #pragma unroll
        for (int j4 = 0; j4 < 8; ++j4) {
            const int c = ocol0 + 4 * j4;
            float4 v = vv[j4];
            if (DO_LN) {
                v.x = (v.x - mu) * rstd * lng[c + 0] + lnb[c + 0];
                v.y = (v.y - mu) * rstd * lng[c + 1] + lnb[c + 1];
                v.z = (v.z - mu) * rstd * lng[c + 2] + lnb[c + 2];
                v.w = (v.w - mu) * rstd * lng[c + 3] + lnb[c + 3];
            }
            *reinterpret_cast<float4*>(yr + c) = v;
        }
    }
}

// ---------------------------------------------------------------------------
// Per-edge fused: eij += Dx[dst] + Ex[src]; sigma = sigmoid(eij);
// atomically scatter sigma*Bx[src] -> num[dst], sigma -> den[dst];
// accumulate per-channel sum / sum-of-squares of eij for edge BatchNorm.
// One wave per edge; lane handles 2 channels. Gathers/atomics hit L2 (25.6MB).
// ---------------------------------------------------------------------------
__global__ __launch_bounds__(256) void k_edge_fuse(
    const float* __restrict__ Dx, const float* __restrict__ Ex,
    const float* __restrict__ Bx,
    const int* __restrict__ srcI, const int* __restrict__ dstI,
    float* __restrict__ eij, float* __restrict__ num, float* __restrict__ den,
    float* __restrict__ gsum, float* __restrict__ gsqs, int E)
{
    __shared__ float esum[64], esqs[64];
    if (threadIdx.x < 64) { esum[threadIdx.x] = 0.f; esqs[threadIdx.x] = 0.f; }
    __syncthreads();

    const int lane = threadIdx.x & 31;
    const int gw = blockIdx.x * 8 + (threadIdx.x >> 5);
    const int nw = gridDim.x * 8;
    const int c = 2 * lane;
    float s0 = 0.f, s1 = 0.f, q0 = 0.f, q1 = 0.f;

    for (int e = gw; e < E; e += nw) {
        const int sn = srcI[e], dn = dstI[e];
        const size_t eo = (size_t)e * 64 + c;
        float2 ce = *reinterpret_cast<float2*>(eij + eo);
        const float2 dv = *reinterpret_cast<const float2*>(Dx + (size_t)dn * 64 + c);
        const float2 ev = *reinterpret_cast<const float2*>(Ex + (size_t)sn * 64 + c);
        const float2 bv = *reinterpret_cast<const float2*>(Bx + (size_t)sn * 64 + c);
        float v0 = ce.x + dv.x + ev.x;
        float v1 = ce.y + dv.y + ev.y;
        float2 ov; ov.x = v0; ov.y = v1;
        *reinterpret_cast<float2*>(eij + eo) = ov;
        s0 += v0; s1 += v1; q0 += v0 * v0; q1 += v1 * v1;
        float g0 = 1.f / (1.f + __expf(-v0));
        float g1 = 1.f / (1.f + __expf(-v1));
        float* np = num + (size_t)dn * 64 + c;
        float* dp = den + (size_t)dn * 64 + c;
        atomicAdd(np,     g0 * bv.x);
        atomicAdd(np + 1, g1 * bv.y);
        atomicAdd(dp,     g0);
        atomicAdd(dp + 1, g1);
    }
    atomicAdd(&esum[c], s0);  atomicAdd(&esum[c + 1], s1);
    atomicAdd(&esqs[c], q0);  atomicAdd(&esqs[c + 1], q1);
    __syncthreads();
    if (threadIdx.x < 64) {
        atomicAdd(&gsum[threadIdx.x], esum[threadIdx.x]);
        atomicAdd(&gsqs[threadIdx.x], esqs[threadIdx.x]);
    }
}

// sums -> per-channel affine (scale, shift) for training-mode BatchNorm
__global__ void k_bn_finalize(const float* __restrict__ gsum,
                              const float* __restrict__ gsqs, float count,
                              const float* __restrict__ g, const float* __restrict__ b,
                              float* __restrict__ scale, float* __restrict__ shift)
{
    int c = threadIdx.x;
    if (c < 64) {
        float m  = gsum[c] / count;
        float v  = gsqs[c] / count - m * m;
        float sc = g[c] * rsqrtf(v + EPS_BN);
        scale[c] = sc;
        shift[c] = b[c] - m * sc;
    }
}

// e += relu(bn(eij)) elementwise (float4 per thread)
__global__ __launch_bounds__(256) void k_edge_apply(
    float* __restrict__ e, const float* __restrict__ eij,
    const float* __restrict__ scale, const float* __restrict__ shift,
    long long total4)
{
    long long i = (long long)blockIdx.x * 256 + threadIdx.x;
    if (i >= total4) return;
    int c = (int)(i & 15) * 4;
    float4 sc = *reinterpret_cast<const float4*>(scale + c);
    float4 sh = *reinterpret_cast<const float4*>(shift + c);
    float4 x  = reinterpret_cast<const float4*>(eij)[i];
    float4 y  = reinterpret_cast<float4*>(e)[i];
    y.x += reluf(x.x * sc.x + sh.x);
    y.y += reluf(x.y * sc.y + sh.y);
    y.z += reluf(x.z * sc.z + sh.z);
    y.w += reluf(x.w * sc.w + sh.w);
    reinterpret_cast<float4*>(e)[i] = y;
}

// xnew = Ax + num/(den+1e-6) in place into Ax, plus node BN channel stats.
// stride is a multiple of 64 -> channel is fixed per thread.
__global__ __launch_bounds__(256) void k_node_combine(
    float* __restrict__ Ax, const float* __restrict__ num,
    const float* __restrict__ den,
    float* __restrict__ gsum, float* __restrict__ gsqs, int total)
{
    __shared__ float ssum[64], ssqs[64];
    if (threadIdx.x < 64) { ssum[threadIdx.x] = 0.f; ssqs[threadIdx.x] = 0.f; }
    __syncthreads();
    const int stride = gridDim.x * 256;
    float s = 0.f, q = 0.f;
    for (int i = blockIdx.x * 256 + threadIdx.x; i < total; i += stride) {
        float v = Ax[i] + num[i] / (den[i] + 1e-6f);
        Ax[i] = v;
        s += v; q += v * v;
    }
    const int c = threadIdx.x & 63;
    atomicAdd(&ssum[c], s);  atomicAdd(&ssqs[c], q);
    __syncthreads();
    if (threadIdx.x < 64) {
        atomicAdd(&gsum[threadIdx.x], ssum[threadIdx.x]);
        atomicAdd(&gsqs[threadIdx.x], ssqs[threadIdx.x]);
    }
}

// x_out = h + relu(bn(xnew)); then LayerNorm(ln2). Wave per node, in-place safe.
__global__ __launch_bounds__(256) void k_node_apply_ln(
    const float* __restrict__ hin, const float* __restrict__ xnew,
    const float* __restrict__ scale, const float* __restrict__ shift,
    const float* __restrict__ lng, const float* __restrict__ lnb,
    float* __restrict__ hout, int N)
{
    const int node = blockIdx.x * 8 + (threadIdx.x >> 5);
    if (node >= N) return;
    const int lane = threadIdx.x & 31;
    const int c = 2 * lane;
    const size_t o = (size_t)node * 64 + c;
    float2 hb = *reinterpret_cast<const float2*>(hin + o);
    float2 xn = *reinterpret_cast<const float2*>(xnew + o);
    float v0 = hb.x + reluf(xn.x * scale[c]     + shift[c]);
    float v1 = hb.y + reluf(xn.y * scale[c + 1] + shift[c + 1]);
    float s = v0 + v1, q = v0 * v0 + v1 * v1;
    #pragma unroll
    for (int off = 16; off > 0; off >>= 1) {
        s += __shfl_xor(s, off, 32);
        q += __shfl_xor(q, off, 32);
    }
    float mu = s * (1.f / 64.f);
    float rstd = rsqrtf(q * (1.f / 64.f) - mu * mu + EPS_LN);
    float2 out;
    out.x = (v0 - mu) * rstd * lng[c]     + lnb[c];
    out.y = (v1 - mu) * rstd * lng[c + 1] + lnb[c + 1];
    *reinterpret_cast<float2*>(hout + o) = out;
}

__global__ __launch_bounds__(256) void k_pool(
    const float* __restrict__ h, const int* __restrict__ batch,
    float* __restrict__ mx, float* __restrict__ sm, int total)
{
    int i = blockIdx.x * 256 + threadIdx.x;
    if (i >= total) return;
    int node = i >> 6, c = i & 63;
    int g = batch[node];
    float v = h[i];
    atomicMaxF(&mx[g * 64 + c], v);
    atomicAdd(&sm[g * 64 + c], v);
}

__global__ void k_pool_accum(float* __restrict__ pooled, const float* __restrict__ mx,
                             const float* __restrict__ sm, const float* __restrict__ cnt)
{
    int i = blockIdx.x * 256 + threadIdx.x;
    if (i >= NGRAPH * 64) return;
    int g = i >> 6, c = i & 63;
    pooled[g * 128 + c]      += mx[i];
    pooled[g * 128 + 64 + c] += sm[i] / fmaxf(cnt[g], 1.f);
}

__global__ void k_hist(const int* __restrict__ batch, float* __restrict__ cnt, int N)
{
    int i = blockIdx.x * 256 + threadIdx.x;
    if (i < N) atomicAdd(&cnt[batch[i]], 1.f);
}

__global__ void k_fill(float* __restrict__ p, float v, int n)
{
    int i = blockIdx.x * 256 + threadIdx.x;
    if (i < n) p[i] = v;
}

// Fused classifier: block per graph row.
// z = relu(pooled @ W1 + b1); z = LN(z, bnl); out = z @ W2 + b2
__global__ __launch_bounds__(256) void k_classifier(
    const float* __restrict__ pooled,
    const float* __restrict__ W1, const float* __restrict__ b1,
    const float* __restrict__ g,  const float* __restrict__ bln,
    const float* __restrict__ W2, const float* __restrict__ b2,
    float* __restrict__ out)
{
    __shared__ float row[128];
    __shared__ float z[256];
    __shared__ float red[256];
    const int tid = threadIdx.x;
    if (tid < 128) row[tid] = pooled[blockIdx.x * 128 + tid];
    __syncthreads();
    float acc = b1[tid];
    #pragma unroll 4
    for (int k = 0; k < 128; ++k) acc += row[k] * W1[k * 256 + tid];
    acc = fmaxf(acc, 0.f);
    red[tid] = acc; __syncthreads();
    for (int s2 = 128; s2 > 0; s2 >>= 1) { if (tid < s2) red[tid] += red[tid + s2]; __syncthreads(); }
    float mean = red[0] * (1.f / 256.f);
    __syncthreads();
    float d = acc - mean;
    red[tid] = d * d; __syncthreads();
    for (int s2 = 128; s2 > 0; s2 >>= 1) { if (tid < s2) red[tid] += red[tid + s2]; __syncthreads(); }
    float var = red[0] * (1.f / 256.f);
    __syncthreads();
    z[tid] = d * rsqrtf(var + EPS_LN) * g[tid] + bln[tid];
    __syncthreads();
    if (tid < 10) {
        float o = b2[tid];
        for (int k = 0; k < 256; ++k) o += z[k] * W2[k * 10 + tid];
        out[blockIdx.x * 10 + tid] = o;
    }
}

// ---------------------------------------------------------------------------
extern "C" void kernel_launch(void* const* d_in, const int* in_sizes, int n_in,
                              void* d_out, int out_size, void* d_ws, size_t ws_size,
                              hipStream_t stream)
{
    const int N = in_sizes[0] / 128;   // 100000 nodes, NODE_F=128
    const int E = in_sizes[2] / 16;    // 800000 edges, EDGE_F=16

    const float* x     = (const float*)d_in[0];
    const int*   eidx  = (const int*)d_in[1];   // [2,E]
    const float* eattr = (const float*)d_in[2];
    const int*   batch = (const int*)d_in[3];
    auto P = [&](int i) { return (const float*)d_in[i]; };
    const float* embW = P(4);
    const float* embB = P(5);
    const int* srcI = eidx;
    const int* dstI = eidx + E;

    // -------- workspace carve (256B-aligned regions) --------
    size_t off = 0;
    auto wsf = [&](size_t nf) {
        float* p = (float*)((char*)d_ws + off);
        off += ((nf * sizeof(float) + 255) & ~(size_t)255);
        return p;
    };
    float* h_a  = wsf((size_t)N * 64);
    float* h_b  = wsf((size_t)N * 64);
    float* Ax   = wsf((size_t)N * 64);   // later holds x_new in place
    float* Bx   = wsf((size_t)N * 64);
    float* Dx   = wsf((size_t)N * 64);
    float* Ex_  = wsf((size_t)N * 64);
    float* num  = wsf((size_t)N * 64);
    float* den  = wsf((size_t)N * 64);
    float* e_cur = wsf((size_t)E * 64);
    float* eij   = wsf((size_t)E * 64);
    float* mx  = wsf(NGRAPH * 64);
    float* sm  = wsf(NGRAPH * 64);
    float* pooled = wsf(NGRAPH * 128);
    float* cnt = wsf(NGRAPH);
    float* gsum = wsf(64);
    float* gsqs = wsf(64);
    float* scl  = wsf(64);
    float* shf  = wsf(64);
    (void)ws_size; (void)n_in; (void)out_size;

    const int gN = (N + 127) / 128;
    const int gE = (E + 127) / 128;

    // -------- per-call init (d_ws is poisoned) --------
    hipMemsetAsync(pooled, 0, NGRAPH * 128 * sizeof(float), stream);
    hipMemsetAsync(cnt, 0, NGRAPH * sizeof(float), stream);
    k_hist<<<(N + 255) / 256, 256, 0, stream>>>(batch, cnt, N);

    // edge embedding: e = edge_attr @ embW + embB   (K=16, zero-padded to 32)
    k_gemm<16, false><<<gE, 256, 0, stream>>>(eattr, E, embW, embB, nullptr, nullptr, e_cur);

    const float* hin = x;  // layer-1 input (K=128)
    for (int l = 0; l < 3; ++l) {
        const int base = 6 + l * 24;
        const float *lt1w = P(base + 0), *lt1b = P(base + 1);
        const float *lt2w = P(base + 2), *lt2b = P(base + 3);
        const float *Aw = P(base + 4),  *Ab = P(base + 5);
        const float *Bw = P(base + 6),  *Bb = P(base + 7);
        const float *Cw = P(base + 8),  *Cb = P(base + 9);
        const float *Dw = P(base + 10), *Db = P(base + 11);
        const float *Ew = P(base + 12), *Eb = P(base + 13);
        const float *bnxg = P(base + 14), *bnxb = P(base + 15);
        const float *bneg = P(base + 16), *bneb = P(base + 17);
        const float *ln1g = P(base + 18), *ln1b = P(base + 19);
        const float *ln2g = P(base + 20), *ln2b = P(base + 21);
        const float *ln3g = P(base + 22), *ln3b = P(base + 23);

        // h_a = LN1(hin @ lt1 + b)
        if (l == 0)
            k_gemm<128, true><<<gN, 256, 0, stream>>>(hin, N, lt1w, lt1b, ln1g, ln1b, h_a);
        else
            k_gemm<64, true><<<gN, 256, 0, stream>>>(hin, N, lt1w, lt1b, ln1g, ln1b, h_a);

        // node projections + edge projection (Ce -> eij, bias included)
        k_gemm<64, false><<<gN, 256, 0, stream>>>(h_a, N, Aw, Ab, nullptr, nullptr, Ax);
        k_gemm<64, false><<<gN, 256, 0, stream>>>(h_a, N, Bw, Bb, nullptr, nullptr, Bx);
        k_gemm<64, false><<<gN, 256, 0, stream>>>(h_a, N, Dw, Db, nullptr, nullptr, Dx);
        k_gemm<64, false><<<gN, 256, 0, stream>>>(h_a, N, Ew, Eb, nullptr, nullptr, Ex_);
        k_gemm<64, false><<<gE, 256, 0, stream>>>(e_cur, E, Cw, Cb, nullptr, nullptr, eij);

        // gather / sigmoid / scatter + edge BN stats
        hipMemsetAsync(num, 0, (size_t)N * 64 * sizeof(float), stream);
        hipMemsetAsync(den, 0, (size_t)N * 64 * sizeof(float), stream);
        hipMemsetAsync(gsum, 0, 64 * sizeof(float), stream);
        hipMemsetAsync(gsqs, 0, 64 * sizeof(float), stream);
        k_edge_fuse<<<2048, 256, 0, stream>>>(Dx, Ex_, Bx, srcI, dstI, eij, num, den,
                                              gsum, gsqs, E);
        k_bn_finalize<<<1, 64, 0, stream>>>(gsum, gsqs, (float)E, bneg, bneb, scl, shf);
        k_edge_apply<<<(int)(((size_t)E * 16 + 255) / 256), 256, 0, stream>>>(
            e_cur, eij, scl, shf, (long long)E * 16);

        // node combine + node BN stats, then residual + BN + LN2 (in place on h_a)
        hipMemsetAsync(gsum, 0, 64 * sizeof(float), stream);
        hipMemsetAsync(gsqs, 0, 64 * sizeof(float), stream);
        k_node_combine<<<4096, 256, 0, stream>>>(Ax, num, den, gsum, gsqs, N * 64);
        k_bn_finalize<<<1, 64, 0, stream>>>(gsum, gsqs, (float)N, bnxg, bnxb, scl, shf);
        k_node_apply_ln<<<(N + 7) / 8, 256, 0, stream>>>(h_a, Ax, scl, shf,
                                                         ln2g, ln2b, h_a, N);

        // h_b = LN3(h_a @ lt2 + b)
        k_gemm<64, true><<<gN, 256, 0, stream>>>(h_a, N, lt2w, lt2b, ln3g, ln3b, h_b);

        // pooled += [segment_max(h_b), segment_mean(h_b)]
        k_fill<<<(NGRAPH * 64 + 255) / 256, 256, 0, stream>>>(mx, -INFINITY, NGRAPH * 64);
        hipMemsetAsync(sm, 0, NGRAPH * 64 * sizeof(float), stream);
        k_pool<<<(N * 64 + 255) / 256, 256, 0, stream>>>(h_b, batch, mx, sm, N * 64);
        k_pool_accum<<<(NGRAPH * 64 + 255) / 256, 256, 0, stream>>>(pooled, mx, sm, cnt);

        hin = h_b;
    }

    // classifier: relu(pooled@W1+b1) -> LN(bnl) -> @W2+b2
    k_classifier<<<NGRAPH, 256, 0, stream>>>(pooled, P(78), P(79), P(80), P(81),
                                             P(82), P(83), (float*)d_out);
}
